// GNNModel_70729521430617
// MI455X (gfx1250) — matrix operand order repaired and checked by reading, not compiled, not run on hardware
//
#include <hip/hip_runtime.h>
#include <hip/hip_bf16.h>

typedef __attribute__((ext_vector_type(16))) _Float16 v16h;
typedef __attribute__((ext_vector_type(8)))  float    v8f;

#define WAVES_PER_BLOCK 4

// ---------------------------------------------------------------------------
// Fused per-node linear + bias + self-loop init:
//   H[i,:]   = act(X)[i,:] @ W^T               (raw messages, gathered later)
//   AGG[i,:] = b + H[i,:] * dinv[i]^2          (bias + self-loop term)
// One wave per 16-node tile, v_wmma_f32_16x16x32_f16, K zero-padded to 32.
// H/AGG/dinv are workspace buffers padded to a multiple of 16 rows, so the
// epilogue needs no bounds checks (padded rows are write-only scratch).
// ---------------------------------------------------------------------------
template<int CIN, int COUT, bool RELU>
__global__ __launch_bounds__(128) void gcn_linear_wmma(
    const float* __restrict__ X, const float* __restrict__ W,
    const float* __restrict__ dinv, const float* __restrict__ bias,
    float* __restrict__ H, float* __restrict__ AGG, int n_nodes)
{
    constexpr int KPAD = (CIN + 31) & ~31;   // 32 or 64
    constexpr int NT   = COUT / 16;          // N tiles of 16
    constexpr int KT   = KPAD / 32;          // K tiles of 32

    __shared__ _Float16 Ws[COUT][KPAD];                  // W row-major, K padded
    __shared__ _Float16 Xs[WAVES_PER_BLOCK][16][KPAD];   // A staging per wave

    const int lane = threadIdx.x & 31;
    const int wave = threadIdx.x >> 5;

    // Stage W (zero-padded in K) into LDS as f16, [n][k] so B-fragment reads
    // are contiguous ds_load_b128.
    for (int idx = threadIdx.x; idx < COUT * KPAD; idx += blockDim.x) {
        const int n = idx / KPAD, k = idx % KPAD;
        Ws[n][k] = (_Float16)((k < CIN) ? W[n * CIN + k] : 0.0f);
    }

    // Stage this wave's 16-node activation tile (optionally ReLU'd) as f16.
    // X is the only buffer we don't own, so guard its reads; zeros elsewhere.
    const int tile = blockIdx.x * WAVES_PER_BLOCK + wave;
    const int row0 = tile * 16;
    for (int idx = lane; idx < 16 * KPAD; idx += 32) {
        const int m = idx / KPAD, k = idx % KPAD;
        const int r = row0 + m;
        float v = 0.0f;
        if (k < CIN && r < n_nodes) {
            v = X[(size_t)r * CIN + k];
            if (RELU) v = fmaxf(v, 0.0f);
        }
        Xs[wave][m][k] = (_Float16)v;
    }
    __syncthreads();

    const int mlo = lane & 15;
    const int hi  = lane >> 4;               // 0: lanes 0-15, 1: lanes 16-31

    // A fragments (ISA 7.12.2, 16-bit A 16x32): K = h + 8*(h>=8) + 8*(lane>=16)
    v16h afrag[KT];
    #pragma unroll
    for (int kt = 0; kt < KT; ++kt) {
        #pragma unroll
        for (int h = 0; h < 16; ++h) {
            const int k = kt * 32 + h + (h >= 8 ? 8 : 0) + (hi ? 8 : 0);
            afrag[kt][h] = Xs[wave][mlo][k];
        }
    }

    // Per-lane dinv^2 for the 8 rows this lane's D fragment covers
    // (contiguous, unconditional: dinv is padded workspace).
    const int rbase = row0 + (hi ? 8 : 0);
    float di2[8];
    #pragma unroll
    for (int r = 0; r < 8; ++r) {
        const float d = dinv[rbase + r];
        di2[r] = d * d;
    }

    #pragma unroll
    for (int nt = 0; nt < NT; ++nt) {
        v8f c = {};
        #pragma unroll
        for (int kt = 0; kt < KT; ++kt) {
            // B fragment (32x16): lane holds N=lane&15, K = h + 8*(h>=8) + 8*(lane>=16)
            v16h bfrag;
            #pragma unroll
            for (int h = 0; h < 16; ++h) {
                const int k = kt * 32 + h + (h >= 8 ? 8 : 0) + (hi ? 8 : 0);
                bfrag[h] = Ws[nt * 16 + mlo][k];
            }
            c = __builtin_amdgcn_wmma_f32_16x16x32_f16(
                    false, afrag[kt], false, bfrag, (short)0, c, false, false);
        }
        // Straight-line epilogue: C/D layout M = r + 8*hi, N = lane&15.
        const int   col = nt * 16 + mlo;
        const float bc  = bias[col];
        #pragma unroll
        for (int r = 0; r < 8; ++r) {
            const size_t o = (size_t)(rbase + r) * COUT + col;
            const float  v = c[r];
            H[o]   = v;
            AGG[o] = bc + v * di2[r];
        }
    }
}

// ---------------------------------------------------------------------------
// Degree / normalization
// ---------------------------------------------------------------------------
__global__ void deg_init_kernel(float* deg, int n) {
    int i = blockIdx.x * blockDim.x + threadIdx.x;
    if (i < n) deg[i] = 1.0f;                     // self-loop contributes 1
}
__global__ void deg_count_kernel(const int* __restrict__ dst, float* deg, int ne) {
    int e = blockIdx.x * blockDim.x + threadIdx.x;
    if (e < ne) atomicAdd(&deg[dst[e]], 1.0f);
}
__global__ void deg_rsqrt_kernel(float* deg, int n) {
    int i = blockIdx.x * blockDim.x + threadIdx.x;
    if (i < n) { float d = deg[i]; deg[i] = (d > 0.0f) ? rsqrtf(d) : 0.0f; }
}

// ---------------------------------------------------------------------------
// agg[dst,:] += h[src,:] * dinv[src]*dinv[dst]  (float4 loads, f32 atomics;
// h and agg fit in the 192MB L2 so gathers/atomics stay on-chip)
// ---------------------------------------------------------------------------
template<int C>
__global__ void edge_scatter_kernel(const int* __restrict__ src,
                                    const int* __restrict__ dst,
                                    const float* __restrict__ dinv,
                                    const float* __restrict__ h,
                                    float* __restrict__ agg, int ne) {
    constexpr int QC = C / 4;
    const long long total = (long long)ne * QC;
    for (long long t = (long long)blockIdx.x * blockDim.x + threadIdx.x; t < total;
         t += (long long)gridDim.x * blockDim.x) {
        const int e = (int)(t / QC);
        const int q = (int)(t % QC);
        const int s = src[e], d = dst[e];
        const float w = dinv[s] * dinv[d];
        const float4 hv = *(const float4*)(h + (size_t)s * C + q * 4);
        float* o = agg + (size_t)d * C + q * 4;
        atomicAdd(o + 0, hv.x * w);
        atomicAdd(o + 1, hv.y * w);
        atomicAdd(o + 2, hv.z * w);
        atomicAdd(o + 3, hv.w * w);
    }
}

// ---------------------------------------------------------------------------
// Pooling + FC head
// ---------------------------------------------------------------------------
__global__ void zero_kernel(float* p, int n) {
    int i = blockIdx.x * blockDim.x + threadIdx.x;
    if (i < n) p[i] = 0.0f;
}
__global__ void pool_accum_kernel(const float* __restrict__ h,
                                  const int* __restrict__ batch,
                                  float* __restrict__ sums,
                                  float* __restrict__ cnts, int n) {
    long long t = (long long)blockIdx.x * blockDim.x + threadIdx.x;
    const long long total = (long long)n * 32;
    if (t < total) {
        const int i = (int)(t >> 5), c = (int)(t & 31);
        const int g = batch[i];
        atomicAdd(&sums[(size_t)g * 32 + c], fmaxf(h[t], 0.0f));   // ReLU of conv3
        if (c == 0) atomicAdd(&cnts[g], 1.0f);
    }
}
__global__ void pool_fc_kernel(const float* __restrict__ sums,
                               const float* __restrict__ cnts,
                               const float* __restrict__ Wfc,
                               const float* __restrict__ bfc,
                               float* __restrict__ out, int ngraphs) {
    int g = blockIdx.x * blockDim.x + threadIdx.x;
    if (g < ngraphs) {
        const float inv = 1.0f / fmaxf(cnts[g], 1.0f);
        float acc = 0.0f;
        #pragma unroll
        for (int c = 0; c < 32; ++c) acc += sums[(size_t)g * 32 + c] * Wfc[c];
        out[g] = acc * inv + bfc[0];
    }
}

// ---------------------------------------------------------------------------
extern "C" void kernel_launch(void* const* d_in, const int* in_sizes, int n_in,
                              void* d_out, int out_size, void* d_ws, size_t ws_size,
                              hipStream_t stream) {
    (void)n_in; (void)ws_size;
    const float* x   = (const float*)d_in[0];
    const int*   ei  = (const int*)d_in[1];
    const int*   bat = (const int*)d_in[2];
    const float* W1  = (const float*)d_in[3];
    const float* b1  = (const float*)d_in[4];
    const float* W2  = (const float*)d_in[5];
    const float* b2  = (const float*)d_in[6];
    const float* W3  = (const float*)d_in[7];
    const float* b3  = (const float*)d_in[8];
    const float* Wfc = (const float*)d_in[9];
    const float* bfc = (const float*)d_in[10];
    float* out = (float*)d_out;

    const int N    = in_sizes[0] / 2;    // x is [N, 2]
    const int E    = in_sizes[1] / 2;    // edge_index is [2, E]
    const int G    = out_size;           // graphs (out is [G,1])
    const int Npad = (N + 15) & ~15;     // 16-row padded buffers -> no GEMM tails
    const int* src = ei;
    const int* dst = ei + E;

    // Workspace carve-out (256B aligned). Triple-buffered activations so the
    // fused gemm can read the previous layer's agg while writing the next.
    size_t off = 0;
    auto alloc = [&](size_t nf) {
        float* p = (float*)((char*)d_ws + off);
        off += ((nf * sizeof(float) + 255) / 256) * 256;
        return p;
    };
    float* dinv = alloc((size_t)Npad);        // degree -> rsqrt in place
    float* bufH = alloc((size_t)Npad * 64);   // raw linear output (messages)
    float* aggA = alloc((size_t)Npad * 64);   // aggregated output (ping)
    float* aggB = alloc((size_t)Npad * 64);   // aggregated output (pong)
    float* sums = alloc((size_t)G * 32);
    float* cnts = alloc((size_t)G);

    const int T = 256;
    auto cdiv = [](long long a, long long b) { return (int)((a + b - 1) / b); };

    // symmetric normalization: deg over dst incl. self-loops, then rsqrt
    // (init over Npad so padded dinv rows are defined, not poison-dependent)
    deg_init_kernel <<<cdiv(Npad, T), T, 0, stream>>>(dinv, Npad);
    deg_count_kernel<<<cdiv(E, T), T, 0, stream>>>(dst, dinv, E);
    deg_rsqrt_kernel<<<cdiv(Npad, T), T, 0, stream>>>(dinv, Npad);

    const int tiles   = Npad / 16;
    const int gblocks = (tiles + WAVES_PER_BLOCK - 1) / WAVES_PER_BLOCK;

    // ---- layer 1: 2 -> 32 ----
    gcn_linear_wmma<2, 32, false><<<gblocks, 128, 0, stream>>>(x, W1, dinv, b1, bufH, aggA, N);
    edge_scatter_kernel<32><<<cdiv((long long)E * 8, T), T, 0, stream>>>(src, dst, dinv, bufH, aggA, E);

    // ---- layer 2: 32 -> 64 (ReLU applied on load) ----
    gcn_linear_wmma<32, 64, true><<<gblocks, 128, 0, stream>>>(aggA, W2, dinv, b2, bufH, aggB, N);
    edge_scatter_kernel<64><<<cdiv((long long)E * 16, T), T, 0, stream>>>(src, dst, dinv, bufH, aggB, E);

    // ---- layer 3: 64 -> 32 ----
    gcn_linear_wmma<64, 32, true><<<gblocks, 128, 0, stream>>>(aggB, W3, dinv, b3, bufH, aggA, N);
    edge_scatter_kernel<32><<<cdiv((long long)E * 8, T), T, 0, stream>>>(src, dst, dinv, bufH, aggA, E);

    // ---- global mean pool (ReLU on read) + FC ----
    zero_kernel<<<cdiv((long long)G * 32, T), T, 0, stream>>>(sums, G * 32);
    zero_kernel<<<cdiv(G, T), T, 0, stream>>>(cnts, G);
    pool_accum_kernel<<<cdiv((long long)N * 32, T), T, 0, stream>>>(aggA, bat, sums, cnts, N);
    pool_fc_kernel<<<cdiv(G, T), T, 0, stream>>>(sums, cnts, Wfc, bfc, out, G);
}
